// SelfAttention_33139967656475
// MI455X (gfx1250) — compile-verified, matrix operand order checked by reading, not run on hardware
//
#include <hip/hip_runtime.h>

#define DEV __device__ __forceinline__

typedef __attribute__((ext_vector_type(16))) __bf16 v16bf;
typedef __attribute__((ext_vector_type(8)))  float  v8f;

// Problem constants (match reference)
// D_MODEL=1024, NUM_HEAD=16, HEAD_DIM=64, BATCH=2, SEQ=2048, M = B*S = 4096

DEV v8f wmma_bf16(v16bf a, v16bf b, v8f c) {
  // D = A(16x32) * B(32x16) + C, f32 accumulate
  return __builtin_amdgcn_wmma_f32_16x16x32_bf16(
      /*neg_a=*/false, a, /*neg_b=*/false, b,
      /*c_mod=*/(short)0, c, /*reuse_a=*/false, /*reuse_b=*/false);
}

// Low 32 bits of a generic pointer into __shared__ memory are the LDS byte
// address (flat->LDS conversion truncates to addr[31:0], ISA 10.2).
DEV unsigned int lds_addr32(const void* p) {
  return (unsigned int)(unsigned long long)(uintptr_t)p;
}

// Async copy 16 bytes global -> LDS (GLOBAL_LOAD_ASYNC_TO_LDS_B128, ASYNCcnt).
// No VGPR data movement; per-lane global address + per-lane LDS address.
DEV void async_b128(unsigned int lds_byte_addr, const void* gaddr) {
  asm volatile("global_load_async_to_lds_b128 %0, %1, off"
               :: "v"(lds_byte_addr),
                  "v"((unsigned long long)(uintptr_t)gaddr)
               : "memory");
}

DEV void wait_async0() {
#if __has_builtin(__builtin_amdgcn_s_wait_asynccnt)
  __builtin_amdgcn_s_wait_asynccnt(0);
#else
  asm volatile("s_wait_asynccnt 0x0" ::: "memory");
#endif
}

// Load one 16x32 bf16 fragment (A-layout; identical addressing works for the
// B operand when the source tile is stored row-major [16 rows][K]):
//   lane<16 : row=lane,    K chunks {0..7, 16..23}
//   lane>=16: row=lane-16, K chunks {8..15, 24..31}
DEV v16bf load_frag16x32(const __bf16* p, int ld) {
  const int lane = threadIdx.x & 31;
  const __bf16* q = p + (lane & 15) * ld + ((lane & 16) ? 8 : 0);
  union { v16bf v; uint4 u[2]; } r;
  r.u[0] = *reinterpret_cast<const uint4*>(q);
  r.u[1] = *reinterpret_cast<const uint4*>(q + 16);
  return r.v;
}

// ---------------------------------------------------------------------------
// f32 -> bf16 conversion (4 elements / thread)
// ---------------------------------------------------------------------------
__global__ void cvt_f32_to_bf16(const float* __restrict__ in,
                                __bf16* __restrict__ out, int n4) {
  int i = blockIdx.x * blockDim.x + threadIdx.x;
  if (i >= n4) return;
  float4 f = reinterpret_cast<const float4*>(in)[i];
  union { __bf16 h[4]; uint2 u; } p;
  p.h[0] = (__bf16)f.x; p.h[1] = (__bf16)f.y;
  p.h[2] = (__bf16)f.z; p.h[3] = (__bf16)f.w;
  reinterpret_cast<uint2*>(out)[i] = p.u;
}

// ---------------------------------------------------------------------------
// Y = X @ W^T + bias. X:[4096x1024] bf16, W:[1024x1024] bf16 (row n = out col).
// MODE 0: write bf16 into [B=2, H=16, S=2048, Dh=64]  (QKV projections)
// MODE 1: write f32 into row-major [4096 x 1024]      (output projection)
// Block: 128 threads (4 waves), 64x64 output tile, K staged 32 at a time via
// async global->LDS copies (ASYNCcnt), fragments read back with ds_load_b128.
// ---------------------------------------------------------------------------
template <int MODE>
__global__ __launch_bounds__(128) void gemm_xwT(
    const __bf16* __restrict__ X, const __bf16* __restrict__ W,
    const float* __restrict__ bias, void* __restrict__ outp) {
  __shared__ __bf16 ldsA[64 * 32];
  __shared__ __bf16 ldsB[64 * 32];

  const int tid  = threadIdx.x;
  const int wave = tid >> 5;
  const int lane = tid & 31;
  const int m0 = blockIdx.x * 64;
  const int n0 = blockIdx.y * 64;

  v8f z = {};
  v8f acc[4] = {z, z, z, z};

  const int ldRow  = tid >> 1;        // 0..63
  const int ldHalf = (tid & 1) * 16;  // 0 or 16 (elements)

  const unsigned int la0 = lds_addr32(&ldsA[ldRow * 32 + ldHalf]);
  const unsigned int la1 = lds_addr32(&ldsA[ldRow * 32 + ldHalf + 8]);
  const unsigned int lb0 = lds_addr32(&ldsB[ldRow * 32 + ldHalf]);
  const unsigned int lb1 = lds_addr32(&ldsB[ldRow * 32 + ldHalf + 8]);

  for (int kk = 0; kk < 1024; kk += 32) {
    __syncthreads();  // previous iteration's LDS reads complete
    const __bf16* gA = X + (m0 + ldRow) * 1024 + kk + ldHalf;
    const __bf16* gB = W + (n0 + ldRow) * 1024 + kk + ldHalf;
    async_b128(la0, gA);
    async_b128(la1, gA + 8);
    async_b128(lb0, gB);
    async_b128(lb1, gB + 8);
    wait_async0();
    __syncthreads();

    v16bf af = load_frag16x32(&ldsA[wave * 16 * 32], 32);
#pragma unroll
    for (int ns = 0; ns < 4; ++ns) {
      v16bf bf = load_frag16x32(&ldsB[ns * 16 * 32], 32);
      acc[ns] = wmma_bf16(af, bf, acc[ns]);
    }
  }

  // Epilogue. C/D layout: VGPR i -> (M = i + (lane>=16 ? 8:0), N = lane&15)
  const int rowAdd = (lane & 16) ? 8 : 0;
  const int nl = lane & 15;
#pragma unroll
  for (int ns = 0; ns < 4; ++ns) {
    const int n = n0 + ns * 16 + nl;
    const float bv = bias[n];
#pragma unroll
    for (int i = 0; i < 8; ++i) {
      const int m = m0 + wave * 16 + i + rowAdd;
      const float v = acc[ns][i] + bv;
      if (MODE == 0) {
        const int bb = m >> 11, s = m & 2047, h = n >> 6, dh = n & 63;
        reinterpret_cast<__bf16*>(outp)[(((bb * 16 + h) * 2048 + s) * 64) + dh] = (__bf16)v;
      } else {
        reinterpret_cast<float*>(outp)[(long)m * 1024 + n] = v;
      }
    }
  }
}

// ---------------------------------------------------------------------------
// Flash attention: one block per (q-tile of 64, head, batch), 4 waves.
// Each wave owns 16 query rows; loops over kv chunks of 64 keys with online
// softmax. K chunk staged via async global->LDS; V staged transposed through
// VGPRs; next chunk prefetched (global_prefetch_b8).
// Q/K/V are bf16 [B,H,S,Dh]; output written bf16 [B,S,H*Dh].
// ---------------------------------------------------------------------------
__global__ __launch_bounds__(128) void attn_fwd(
    const __bf16* __restrict__ Qb, const __bf16* __restrict__ Kb,
    const __bf16* __restrict__ Vb, const float* __restrict__ mask,
    __bf16* __restrict__ Ob) {
  __shared__ __bf16 ldsK[64 * 64];        // [key][dh]
  __shared__ __bf16 ldsVt[64 * 64];       // [dh][key]
  __shared__ __bf16 ldsP[4 * 16 * 64];    // per-wave P tile [16 q][64 key]

  const int tid = threadIdx.x, wave = tid >> 5, lane = tid & 31;
  const int h = blockIdx.y, b = blockIdx.z;
  const long head = (long)(b * 16 + h) * 2048 * 64;
  const __bf16* Qh = Qb + head;
  const __bf16* Kh = Kb + head;
  const __bf16* Vh = Vb + head;

  const int q0 = blockIdx.x * 64 + wave * 16;  // first q row for this wave

  // Q strip 16x64 -> two A fragments, pinned in registers
  const v16bf qf0 = load_frag16x32(Qh + (long)q0 * 64 + 0, 64);
  const v16bf qf1 = load_frag16x32(Qh + (long)q0 * 64 + 32, 64);

  v8f z = {};
  v8f oacc[4] = {z, z, z, z};
  float mrow[8], lrow[8];
#pragma unroll
  for (int i = 0; i < 8; ++i) { mrow[i] = -__builtin_huge_valf(); lrow[i] = 0.f; }

  const int rowAdd = (lane & 16) ? 8 : 0;
  const int nl = lane & 15;
  const int ldRow = tid >> 1;        // 0..63
  const int ldCol = (tid & 1) * 32;  // 0 or 32

  unsigned int lk[4];
#pragma unroll
  for (int ii = 0; ii < 4; ++ii)
    lk[ii] = lds_addr32(&ldsK[ldRow * 64 + ldCol + ii * 8]);

  for (int j = 0; j < 2048; j += 64) {
    __syncthreads();
    // Stage K chunk [64 key][64 dh] asynchronously (no VGPR round-trip)
    const __bf16* gK = Kh + (long)(j + ldRow) * 64 + ldCol;
    const __bf16* gV = Vh + (long)(j + ldRow) * 64 + ldCol;
#pragma unroll
    for (int ii = 0; ii < 4; ++ii) async_b128(lk[ii], gK + ii * 8);

    // Stage V chunk transposed: ldsVt[dh][key]
#pragma unroll
    for (int ii = 0; ii < 4; ++ii) {
      union { uint4 u; __bf16 hh[8]; } t;
      t.u = *reinterpret_cast<const uint4*>(gV + ii * 8);
#pragma unroll
      for (int e = 0; e < 8; ++e)
        ldsVt[(ldCol + ii * 8 + e) * 64 + ldRow] = t.hh[e];
    }

    // Prefetch next chunk while this one is consumed
    if (j + 64 < 2048) {
      __builtin_prefetch(gK + 64 * 64, 0, 1);
      __builtin_prefetch(gV + 64 * 64, 0, 1);
    }

    wait_async0();
    __syncthreads();

    // scores = Q @ K^T  (16 x 64)
    v8f sacc[4];
#pragma unroll
    for (int ns = 0; ns < 4; ++ns) {
      v16bf kf0 = load_frag16x32(&ldsK[ns * 16 * 64 + 0], 64);
      v16bf kf1 = load_frag16x32(&ldsK[ns * 16 * 64 + 32], 64);
      v8f sa = z;
      sa = wmma_bf16(qf0, kf0, sa);
      sa = wmma_bf16(qf1, kf1, sa);
      sacc[ns] = sa;
    }

    // scale + mask
#pragma unroll
    for (int ns = 0; ns < 4; ++ns)
#pragma unroll
      for (int i = 0; i < 8; ++i) {
        const int qg = q0 + i + rowAdd;
        const int kg = j + ns * 16 + nl;
        sacc[ns][i] = sacc[ns][i] * 0.125f + mask[(long)qg * 2048 + kg];
      }

    // online softmax per query row; P -> per-wave LDS tile (bf16, A-layout src)
    __bf16* Pw = &ldsP[wave * 16 * 64];
#pragma unroll
    for (int i = 0; i < 8; ++i) {
      float mx = sacc[0][i];
      mx = fmaxf(mx, sacc[1][i]);
      mx = fmaxf(mx, sacc[2][i]);
      mx = fmaxf(mx, sacc[3][i]);
#pragma unroll
      for (int d = 8; d >= 1; d >>= 1) mx = fmaxf(mx, __shfl_xor(mx, d, 32));
      const float mnew = fmaxf(mrow[i], mx);
      const float alpha = __expf(mrow[i] - mnew);
      mrow[i] = mnew;
      lrow[i] *= alpha;
      float ps = 0.f;
#pragma unroll
      for (int ns = 0; ns < 4; ++ns) {
        oacc[ns][i] *= alpha;
        const float p = __expf(sacc[ns][i] - mnew);
        ps += p;
        Pw[(i + rowAdd) * 64 + ns * 16 + nl] = (__bf16)p;
      }
#pragma unroll
      for (int d = 8; d >= 1; d >>= 1) ps += __shfl_xor(ps, d, 32);
      lrow[i] += ps;
    }
    // Make the type-punned LDS reads below observe the bf16 stores above
    // (same wave, so no barrier needed; just block compiler reordering).
    asm volatile("" ::: "memory");

    // O += P @ V
    const v16bf pf0 = load_frag16x32(Pw + 0, 64);
    const v16bf pf1 = load_frag16x32(Pw + 32, 64);
#pragma unroll
    for (int ns = 0; ns < 4; ++ns) {
      v16bf vf0 = load_frag16x32(&ldsVt[ns * 16 * 64 + 0], 64);
      v16bf vf1 = load_frag16x32(&ldsVt[ns * 16 * 64 + 32], 64);
      oacc[ns] = wmma_bf16(pf0, vf0, oacc[ns]);
      oacc[ns] = wmma_bf16(pf1, vf1, oacc[ns]);
    }
  }

  // Normalize and write concat layout [B, S, H*Dh] as bf16
#pragma unroll
  for (int ns = 0; ns < 4; ++ns)
#pragma unroll
    for (int i = 0; i < 8; ++i) {
      const int qg = q0 + i + rowAdd;
      const int dh = ns * 16 + nl;
      const float v = oacc[ns][i] / lrow[i];
      Ob[((long)(b * 2048 + qg)) * 1024 + h * 64 + dh] = (__bf16)v;
    }
}

// ---------------------------------------------------------------------------
extern "C" void kernel_launch(void* const* d_in, const int* in_sizes, int n_in,
                              void* d_out, int out_size, void* d_ws, size_t ws_size,
                              hipStream_t stream) {
  (void)in_sizes; (void)n_in; (void)out_size; (void)ws_size;
  const float* x    = (const float*)d_in[0];
  const float* mask = (const float*)d_in[1];
  const float* Wq   = (const float*)d_in[2];
  const float* bq   = (const float*)d_in[3];
  const float* Wk   = (const float*)d_in[4];
  const float* bk   = (const float*)d_in[5];
  const float* Wv   = (const float*)d_in[6];
  const float* bv   = (const float*)d_in[7];
  const float* Wo   = (const float*)d_in[8];
  const float* bo   = (const float*)d_in[9];
  float* out = (float*)d_out;

  // Workspace partition (bf16 staging buffers), 256B aligned slices.
  char* ws = (char*)d_ws;
  size_t off = 0;
  auto alloc = [&](size_t bytes) {
    char* p = ws + off;
    off += (bytes + 255) & ~(size_t)255;
    return p;
  };
  __bf16* xb  = (__bf16*)alloc((size_t)4096 * 1024 * 2);
  __bf16* wqb = (__bf16*)alloc((size_t)1024 * 1024 * 2);
  __bf16* wkb = (__bf16*)alloc((size_t)1024 * 1024 * 2);
  __bf16* wvb = (__bf16*)alloc((size_t)1024 * 1024 * 2);
  __bf16* wob = (__bf16*)alloc((size_t)1024 * 1024 * 2);
  __bf16* qb  = (__bf16*)alloc((size_t)4096 * 1024 * 2);
  __bf16* kb  = (__bf16*)alloc((size_t)4096 * 1024 * 2);
  __bf16* vb  = (__bf16*)alloc((size_t)4096 * 1024 * 2);
  __bf16* ab  = (__bf16*)alloc((size_t)4096 * 1024 * 2);

  // 1) f32 -> bf16 conversions
  {
    int n4 = (4096 * 1024) / 4;
    cvt_f32_to_bf16<<<(n4 + 255) / 256, 256, 0, stream>>>(x, xb, n4);
    n4 = (1024 * 1024) / 4;
    cvt_f32_to_bf16<<<(n4 + 255) / 256, 256, 0, stream>>>(Wq, wqb, n4);
    cvt_f32_to_bf16<<<(n4 + 255) / 256, 256, 0, stream>>>(Wk, wkb, n4);
    cvt_f32_to_bf16<<<(n4 + 255) / 256, 256, 0, stream>>>(Wv, wvb, n4);
    cvt_f32_to_bf16<<<(n4 + 255) / 256, 256, 0, stream>>>(Wo, wob, n4);
  }

  const dim3 gGemm(64, 16), blk(128);
  // 2) Q/K/V projections (bf16 out in [B,H,S,Dh])
  gemm_xwT<0><<<gGemm, blk, 0, stream>>>(xb, wqb, bq, qb);
  gemm_xwT<0><<<gGemm, blk, 0, stream>>>(xb, wkb, bk, kb);
  gemm_xwT<0><<<gGemm, blk, 0, stream>>>(xb, wvb, bv, vb);

  // 3) Flash attention -> concat [B,S,D] bf16
  attn_fwd<<<dim3(32, 16, 2), blk, 0, stream>>>(qb, kb, vb, mask, ab);

  // 4) Output projection (f32 out = d_out)
  gemm_xwT<1><<<gGemm, blk, 0, stream>>>(ab, wob, bo, (void*)out);
}
